// DoubleSubstitutionEmbedding_10471130267741
// MI455X (gfx1250) — matrix operand order repaired and checked by reading, not compiled, not run on hardware
//
#include <hip/hip_runtime.h>
#include <hip/hip_bf16.h>
#include <stdint.h>

// ---------------- CDNA5 (gfx1250, wave32) WMMA GEMM pipeline ------------------
// embed(f32 tables -> f16) -> GEMM0 (v_wmma_f32_16x16x32_f16, async-LDS double
// buffer) -> masked scatter (rank/compaction scan) -> GEMM1 -> scatter -> GEMM2.

typedef __attribute__((ext_vector_type(16))) _Float16 v16h;
typedef __attribute__((ext_vector_type(8)))  _Float16 v8h;
typedef __attribute__((ext_vector_type(8)))  float    v8f;

#define BB     8
#define L2L    1024
#define L1L    4096
#define L0L    16384
#define STOT   (L2L + L1L + L0L)   // 21504
#define RES2   64                  // 2 * RESOLUTION
#define SPD    3

// ---------------------------------------------------------------------------
// Embedding: x[b,t,c] = vtab[value][c] + sum_a ptab[a][pos_a][c], cast to f16.
// ---------------------------------------------------------------------------
__global__ __launch_bounds__(256)
void embed_kernel(const int* __restrict__ value, const int* __restrict__ position,
                  const float* __restrict__ vtab, const float* __restrict__ ptab,
                  _Float16* __restrict__ x, int L, int sOff, int e)
{
    size_t i = (size_t)blockIdx.x * blockDim.x + threadIdx.x;
    size_t total = (size_t)BB * L * e;
    if (i >= total) return;
    int    c   = (int)(i % e);
    size_t row = i / e;                 // b*L + t
    int b = (int)(row / L);
    int t = (int)(row % L);
    size_t s = (size_t)b * STOT + sOff + t;
    int v = value[s];
    if (v < 0) v = 0; if (v > 3) v = 3;
    float acc = vtab[(size_t)v * e + c];
#pragma unroll
    for (int a = 0; a < SPD; ++a) {
        int p = position[s * SPD + a];
        if (p < 0) p = 0; if (p >= RES2) p = RES2 - 1;
        acc += ptab[((size_t)a * RES2 + p) * e + c];
    }
    x[i] = (_Float16)acc;
}

// ---------------------------------------------------------------------------
// Weight transform: conv_w [O, C, Kk] f32 -> Wt f16 [O, Kk*C] (= W^T with the
// GEMM reduction index kk*C + c). Output row o is contiguous in K.
// ---------------------------------------------------------------------------
__global__ __launch_bounds__(256)
void wtrans_kernel(const float* __restrict__ w, _Float16* __restrict__ wt,
                   int O, int C, int Kk)
{
    size_t i = (size_t)blockIdx.x * blockDim.x + threadIdx.x;
    size_t total = (size_t)O * C * Kk;
    if (i >= total) return;
    int per = C * Kk;
    int o   = (int)(i / per);
    int rem = (int)(i % per);
    int kk  = rem / C;
    int c   = rem % C;
    wt[i] = (_Float16)w[(size_t)o * per + (size_t)c * Kk + kk];
}

// ---------------------------------------------------------------------------
// Tiled WMMA GEMM:  C[M,N](f32) = A[M,K](f16) * Bt[N,K]^T(f16) + bias[N]
// 128 threads = 4 waves; 64x64 block tile; each wave owns 32x32 (2x2 WMMA
// accumulators). K stepped by 32 through double-buffered LDS, staged with
// GLOBAL_LOAD_ASYNC_TO_LDS_B128 (ASYNCcnt-tracked DMA, no VGPR round-trip).
// ---------------------------------------------------------------------------
__device__ __forceinline__ void async_tile_copy(uint32_t lds_off, const _Float16* g)
{
    // 32 bytes per thread: two b128 async copies; offset applies to both sides.
    asm volatile("global_load_async_to_lds_b128 %0, %1, off"
                 :: "v"(lds_off), "v"(g) : "memory");
    asm volatile("global_load_async_to_lds_b128 %0, %1, off offset:16"
                 :: "v"(lds_off), "v"(g) : "memory");
}

__global__ __launch_bounds__(128)
void gemm_wmma_kernel(const _Float16* __restrict__ A,
                      const _Float16* __restrict__ Bt,
                      const float* __restrict__ bias,
                      float* __restrict__ C,
                      int M, int N, int K)
{
    __shared__ __align__(16) _Float16 As[2][64][32];
    __shared__ __align__(16) _Float16 Bs[2][64][32];

    const int tid  = threadIdx.x;
    const int lane = tid & 31;
    const int wave = tid >> 5;
    const int m0 = blockIdx.y * 64;
    const int n0 = blockIdx.x * 64;
    const int wm = (wave >> 1) * 32;     // wave's M offset inside block tile
    const int wn = (wave & 1)  * 32;     // wave's N offset inside block tile

    // global->LDS staging: 64 rows x 32 halves (64B/row), 2 threads per row.
    const int lrow = tid >> 1;           // 0..63
    const int lcol = (tid & 1) * 16;     // 0 or 16
    const _Float16* aRow = A  + (size_t)(m0 + lrow) * K + lcol;
    const _Float16* bRow = Bt + (size_t)(n0 + lrow) * K + lcol;

    const int r  = lane & 15;            // row/col within 16x16 tile
    const int hi = lane >> 4;            // lane half selects K split

    v8f acc[2][2] = {};

    auto stage = [&](int buf, int k0) {
        // flat->LDS: low 32 bits of a shared pointer are the LDS byte offset
        uint32_t la = (uint32_t)(uintptr_t)&As[buf][lrow][lcol];
        uint32_t lb = (uint32_t)(uintptr_t)&Bs[buf][lrow][lcol];
        async_tile_copy(la, aRow + k0);
        async_tile_copy(lb, bRow + k0);
    };

    const int nIter = K >> 5;
    stage(0, 0);
    for (int it = 0; it < nIter; ++it) {
        asm volatile("s_wait_asynccnt 0x0" ::: "memory");  // this wave's tile landed
        __syncthreads();                                   // whole block's tile landed
        if (it + 1 < nIter) stage((it + 1) & 1, (it + 1) << 5);
        const int cur = it & 1;

        // A fragment (16-bit A 16x32): lanes 0-15 hold K 0..7 & 16..23,
        // lanes 16-31 hold K 8..15 & 24..31, for row M = lane%16.
        v16h afrag[2], bfrag[2];
#pragma unroll
        for (int mt = 0; mt < 2; ++mt) {
            const _Float16* src = &As[cur][wm + mt * 16 + r][0];
            v8h lo8 = *(const v8h*)(src + hi * 8);
            v8h hi8 = *(const v8h*)(src + 16 + hi * 8);
            v16h a;
#pragma unroll
            for (int q = 0; q < 8; ++q) { a[q] = lo8[q]; a[8 + q] = hi8[q]; }
            afrag[mt] = a;
        }
        // B fragment (32x16): lane holds column N = lane%16; lanes 0-15 carry
        // K 0..7 & 16..23, lanes 16-31 carry K 8..15 & 24..31 (matches A).
#pragma unroll
        for (int nt = 0; nt < 2; ++nt) {
            const _Float16* src = &Bs[cur][wn + nt * 16 + r][0];
            v8h lo8 = *(const v8h*)(src + hi * 8);
            v8h hi8 = *(const v8h*)(src + 16 + hi * 8);
            v16h b;
#pragma unroll
            for (int q = 0; q < 8; ++q) { b[q] = lo8[q]; b[8 + q] = hi8[q]; }
            bfrag[nt] = b;
        }
#pragma unroll
        for (int mt = 0; mt < 2; ++mt)
#pragma unroll
            for (int nt = 0; nt < 2; ++nt)
                acc[mt][nt] = __builtin_amdgcn_wmma_f32_16x16x32_f16(
                    false, afrag[mt], false, bfrag[nt],
                    (short)0, acc[mt][nt], false, false);
    }

    // Epilogue: C/D layout = 8 VGPRs; lane<16: N=lane, M=vgpr; lane>=16: M=vgpr+8.
#pragma unroll
    for (int mt = 0; mt < 2; ++mt) {
#pragma unroll
        for (int nt = 0; nt < 2; ++nt) {
            int n  = n0 + wn + nt * 16 + r;
            int mb = m0 + wm + mt * 16 + hi * 8;
            float bv = bias ? bias[n] : 0.0f;
#pragma unroll
            for (int q = 0; q < 8; ++q)
                C[(size_t)(mb + q) * N + n] = acc[mt][nt][q] + bv;
        }
    }
}

// ---------------------------------------------------------------------------
// Single-block mask scan over B*L elements of the value stream.
//  eqTwo=1: mask = (value == 2), write exclusive rank per element.
//  eqTwo=0: mask = (value != 0), compact True positions into srcpos.
// ---------------------------------------------------------------------------
__global__ __launch_bounds__(1024)
void mask_scan_kernel(const int* __restrict__ value, int L, int sOff, int step,
                      int eqTwo, int* __restrict__ rank,
                      int* __restrict__ srcpos, int* __restrict__ total)
{
    __shared__ int sdata[1024];
    __shared__ int running;
    int tid = threadIdx.x;
    if (tid == 0) running = 0;
    __syncthreads();
    int n = BB * L;
    for (int base = 0; base < n; base += 1024) {
        int i = base + tid;
        int m = 0;
        if (i < n) {
            int b = i / L, j = i % L;
            int v = value[(size_t)b * STOT + sOff + (size_t)j * step];
            m = eqTwo ? (v == 2) : (v != 0);
        }
        sdata[tid] = m;
        __syncthreads();
        for (int off = 1; off < 1024; off <<= 1) {
            int t = (tid >= off) ? sdata[tid - off] : 0;
            __syncthreads();
            sdata[tid] += t;
            __syncthreads();
        }
        int excl = sdata[tid] - m;
        int rk = running + excl;
        if (i < n) {
            if (rank) rank[i] = rk;
            if (srcpos && m) srcpos[rk] = i;
        }
        __syncthreads();
        if (tid == 0) running += sdata[1023];
        __syncthreads();
    }
    if (tid == 0 && total) *total = running;
}

// ---------------------------------------------------------------------------
// Masked scatter: for dst rows with value==2, copy the rank-paired src row of
// y (f32) into the f16 activation buffer x.
// ---------------------------------------------------------------------------
__global__ __launch_bounds__(256)
void scatter_kernel(const float* __restrict__ y, _Float16* __restrict__ x,
                    const int* __restrict__ value, int L, int sOff,
                    const int* __restrict__ rank, const int* __restrict__ srcpos,
                    const int* __restrict__ nsrc, int E, int Msrc)
{
    int row = blockIdx.x;                 // 0 .. BB*L-1
    int b = row / L, j = row % L;
    if (value[(size_t)b * STOT + sOff + j] != 2) return;
    int nt = *nsrc;
    int r  = rank[row];
    int src = 0;
    if (nt > 0) {
        if (r >= nt) r = nt - 1;
        src = srcpos[r];
    }
    if (src >= Msrc) src = Msrc - 1;
    for (int c = threadIdx.x; c < E; c += blockDim.x)
        x[(size_t)row * E + c] = (_Float16)y[(size_t)src * E + c];
}

// ---------------------------------------------------------------------------
extern "C" void kernel_launch(void* const* d_in, const int* in_sizes, int n_in,
                              void* d_out, int out_size, void* d_ws, size_t ws_size,
                              hipStream_t stream)
{
    (void)in_sizes; (void)n_in; (void)out_size; (void)ws_size;
    const int*   value    = (const int*)d_in[0];
    // d_in[1] = depth: layout is structurally fixed (contiguous by depth)
    const int*   position = (const int*)d_in[2];
    const float* e0v = (const float*)d_in[3];
    const float* e0p = (const float*)d_in[4];
    const float* e1v = (const float*)d_in[5];
    const float* e1p = (const float*)d_in[6];
    const float* e2v = (const float*)d_in[7];
    const float* e2p = (const float*)d_in[8];
    const float* w0  = (const float*)d_in[9];
    const float* b0  = (const float*)d_in[10];
    const float* w1  = (const float*)d_in[11];
    const float* b1  = (const float*)d_in[12];
    const float* w2  = (const float*)d_in[13];
    const float* b2  = (const float*)d_in[14];
    float* out = (float*)d_out;

    // ---- workspace layout (byte offsets, 256B aligned) ----
    uint8_t* ws = (uint8_t*)d_ws;
    size_t off = 0;
    auto alloc = [&](size_t bytes) { void* p = ws + off; off += (bytes + 255) & ~(size_t)255; return p; };
    _Float16* x0h = (_Float16*)alloc((size_t)BB * L0L * 128 * 2);  // 33.5 MB
    _Float16* x1h = (_Float16*)alloc((size_t)BB * L1L * 256 * 2);  // 16.8 MB
    _Float16* x2h = (_Float16*)alloc((size_t)BB * L2L * 512 * 2);  //  8.4 MB
    _Float16* wt0 = (_Float16*)alloc((size_t)256  * 1024 * 2);
    _Float16* wt1 = (_Float16*)alloc((size_t)512  * 2048 * 2);
    _Float16* wt2 = (_Float16*)alloc((size_t)1024 * 4096 * 2);
    float* y0 = (float*)alloc((size_t)BB * 2048 * 256 * 4);        // 16.8 MB
    float* y1 = (float*)alloc((size_t)BB * 512  * 512 * 4);        //  8.4 MB
    int* rank1   = (int*)alloc((size_t)BB * L1L * 4);
    int* srcpos0 = (int*)alloc((size_t)BB * 2048 * 4);
    int* nsrc0   = (int*)alloc(4);
    int* rank2   = (int*)alloc((size_t)BB * L2L * 4);
    int* srcpos1 = (int*)alloc((size_t)BB * 512 * 4);
    int* nsrc1   = (int*)alloc(4);

    // ---- embeddings (f32 tables -> f16 activations) ----
    {
        size_t t0 = (size_t)BB * L0L * 128;
        embed_kernel<<<dim3((unsigned)((t0 + 255) / 256)), 256, 0, stream>>>(
            value, position, e0v, e0p, x0h, L0L, L2L + L1L, 128);
        size_t t1 = (size_t)BB * L1L * 256;
        embed_kernel<<<dim3((unsigned)((t1 + 255) / 256)), 256, 0, stream>>>(
            value, position, e1v, e1p, x1h, L1L, L2L, 256);
        size_t t2 = (size_t)BB * L2L * 512;
        embed_kernel<<<dim3((unsigned)((t2 + 255) / 256)), 256, 0, stream>>>(
            value, position, e2v, e2p, x2h, L2L, 0, 512);
    }

    // ---- weight transposes -> f16 [N, K] ----
    wtrans_kernel<<<dim3((256u * 1024u) / 256u),  256, 0, stream>>>(w0, wt0, 256, 128, 8);
    wtrans_kernel<<<dim3((512u * 2048u) / 256u),  256, 0, stream>>>(w1, wt1, 512, 256, 8);
    wtrans_kernel<<<dim3((1024u * 4096u) / 256u), 256, 0, stream>>>(w2, wt2, 1024, 512, 8);

    // ---- scatter index scans (general rank/compaction pairing) ----
    mask_scan_kernel<<<1, 1024, 0, stream>>>(value, L0L / 8, L2L + L1L, 8, 0, nullptr, srcpos0, nsrc0);
    mask_scan_kernel<<<1, 1024, 0, stream>>>(value, L1L,     L2L,      1, 1, rank1,  nullptr, nullptr);
    mask_scan_kernel<<<1, 1024, 0, stream>>>(value, L1L / 8, L2L,      8, 0, nullptr, srcpos1, nsrc1);
    mask_scan_kernel<<<1, 1024, 0, stream>>>(value, L2L,     0,        1, 1, rank2,  nullptr, nullptr);

    // ---- GEMM0: y0[16384,256] = x0[16384,1024] * W0^T + b0 ----
    gemm_wmma_kernel<<<dim3(256 / 64, 16384 / 64), 128, 0, stream>>>(
        x0h, wt0, b0, y0, 16384, 256, 1024);
    // scatter y0 rows into x1 (f16)
    scatter_kernel<<<dim3(BB * L1L), 256, 0, stream>>>(
        y0, x1h, value, L1L, L2L, rank1, srcpos0, nsrc0, 256, BB * 2048);

    // ---- GEMM1: y1[4096,512] = x1[4096,2048] * W1^T + b1 ----
    gemm_wmma_kernel<<<dim3(512 / 64, 4096 / 64), 128, 0, stream>>>(
        x1h, wt1, b1, y1, 4096, 512, 2048);
    scatter_kernel<<<dim3(BB * L2L), 256, 0, stream>>>(
        y1, x2h, value, L2L, 0, rank2, srcpos1, nsrc1, 512, BB * 512);

    // ---- GEMM2: out[1024,1024] = x2[1024,4096] * W2^T + b2 ----
    gemm_wmma_kernel<<<dim3(1024 / 64, 1024 / 64), 128, 0, stream>>>(
        x2h, wt2, b2, out, 1024, 1024, 4096);
}
